// Decoder_17197049053822
// MI455X (gfx1250) — matrix operand order repaired and checked by reading, not compiled, not run on hardware
//
#include <hip/hip_runtime.h>
#include <math.h>

typedef __attribute__((ext_vector_type(16))) __bf16 v16bf;
typedef __attribute__((ext_vector_type(8)))  float  v8f;

__device__ __forceinline__ unsigned short f2bf(float x) {
  unsigned int u = __float_as_uint(x);
  u += 0x7FFFu + ((u >> 16) & 1u);           // round-to-nearest-even
  return (unsigned short)(u >> 16);
}
__device__ __forceinline__ float sigmoidf(float x) { return 1.0f / (1.0f + expf(-x)); }

// ---------------------------------------------------------------------------
// K1: embedding gather + GRU1 + u_pred = pred @ d1_w        (tiny, one block)
// ---------------------------------------------------------------------------
__global__ __launch_bounds__(256) void k1_embed_gru1_upred(
    const int* __restrict__ tokens, const float* __restrict__ emb_table,
    const float* __restrict__ h0, const float* __restrict__ g1k,
    const float* __restrict__ g1r, const float* __restrict__ g1b,
    const float* __restrict__ d1w,
    float* __restrict__ emb_out, float* __restrict__ pred_out,
    float* __restrict__ upred_out)
{
  __shared__ float emb_sm[16 * 256];
  __shared__ float pred_sm[16 * 256];
  const int tid = threadIdx.x;
  for (int o = tid; o < 16 * 256; o += 256) {
    int i = o >> 8, e = o & 255;
    float v = emb_table[tokens[i] * 256 + e];
    emb_sm[o] = v; emb_out[o] = v;
  }
  __syncthreads();
  for (int o = tid; o < 16 * 256; o += 256) {
    int i = o >> 8, j = o & 255;
    float xz = g1b[j], xr = g1b[256 + j], xh = g1b[512 + j];
    float hz = g1b[768 + j], hr = g1b[768 + 256 + j], hh = g1b[768 + 512 + j];
    for (int c = 0; c < 256; ++c) {
      float x = emb_sm[i * 256 + c];
      xz += x * g1k[c * 768 + j];
      xr += x * g1k[c * 768 + 256 + j];
      xh += x * g1k[c * 768 + 512 + j];
      float h = h0[i * 256 + c];
      hz += h * g1r[c * 768 + j];
      hr += h * g1r[c * 768 + 256 + j];
      hh += h * g1r[c * 768 + 512 + j];
    }
    float z = sigmoidf(xz + hz), r = sigmoidf(xr + hr);
    float hc = tanhf(xh + r * hh);
    float p = z * h0[o] + (1.f - z) * hc;
    pred_sm[o] = p; pred_out[o] = p;
  }
  __syncthreads();
  for (int o = tid; o < 16 * 512; o += 256) {
    int i = o >> 9, d = o & 511;
    float s = 0.f;
    for (int j = 0; j < 256; ++j) s += pred_sm[i * 256 + j] * d1w[j * 512 + d];
    upred_out[o] = s;
  }
}

// ---------------------------------------------------------------------------
// Prep: fb[d] = b1 @ w2   (conv bias folded through the 1x1 projection)
// ---------------------------------------------------------------------------
__global__ __launch_bounds__(256) void k_prep_fb(
    const float* __restrict__ b1, const float* __restrict__ w2,
    float* __restrict__ fb)
{
  int d = blockIdx.x * 256 + threadIdx.x;
  if (d >= 512) return;
  float s = 0.f;
  for (int c = 0; c < 256; ++c) s += b1[c] * w2[c * 512 + d];
  fb[d] = s;
}

// ---------------------------------------------------------------------------
// Prep: build the augmented, WMMA-swizzled bf16 B matrix  [KTOT x 512]
//   rows [0,C)      : w3[k,d]
//   rows [C,C+T)    : fw[t,d] = sum_c w1[t,c] * w2[c,d]   (folded conv)
//   rows [C+T,KTOT) : 0
// Swizzle for v_wmma_f32_16x16x32_bf16 B, contiguous k order (same as the A fragment):
//   lane = (k%32)/16*16 + d%16 ; vgpr v = (k%16)/2 ; pair bit = k&1
// ---------------------------------------------------------------------------
__global__ __launch_bounds__(256) void k_prep_B(
    const float* __restrict__ w1, const float* __restrict__ w2,
    const float* __restrict__ w3, unsigned short* __restrict__ Bsw,
    int C, int T, int KTOT)
{
  int idx = blockIdx.x * 256 + threadIdx.x;
  int total = KTOT * 512;
  if (idx >= total) return;
  int k = idx >> 9, d = idx & 511;
  float v = 0.f;
  if (k < C) {
    v = w3[(size_t)k * 512 + d];
  } else {
    int t = k - C;
    if (t < T) {
      float s = 0.f;
      for (int c = 0; c < 256; ++c) s += w1[t * 256 + c] * w2[c * 512 + d];
      v = s;
    }
  }
  int ktile = k >> 5, kr = k & 31;
  int halfk = kr >> 4, kr16 = kr & 15;
  int vv = kr16 >> 1, bit = kr16 & 1;
  int ntile = d >> 4, n = d & 15;
  int lane = (halfk << 4) | n;
  int NK = KTOT >> 5;
  size_t us = ((((size_t)ntile * NK + ktile) * 32 + lane) * 8 + vv) * 2 + bit;
  Bsw[us] = f2bf(v);
}

// ---------------------------------------------------------------------------
// Fused attention-score kernel (the WMMA workhorse):
//   e[p] = sum_d tanh( [inputs(p,:)|taps(p,:)] @ B[:,d] + fb[d] + upred[b,d] ) * w4[d]
// 32 rows per block; 8 waves = 2 row-subtiles x 4 column groups; D = 512.
// All per-column scalars (fb+upred, w4) are preloaded so the unrolled GEMM
// stream is pure { b128-pair clause -> wmma } with no loadcnt drains.
// ---------------------------------------------------------------------------
template<int C, int T, int KW, int PAD, int H, int W, int KTOT>
__global__ __launch_bounds__(256) void k_attn_scores(
    const float* __restrict__ inputs, const float* __restrict__ attn_sum,
    const float* __restrict__ upred, const float* __restrict__ fb,
    const float* __restrict__ w4, const unsigned int* __restrict__ Bsw,
    float* __restrict__ e_out)
{
  constexpr int AST = KTOT + 8;          // LDS row stride (bank-spread pad)
  constexpr int NK  = KTOT / 32;
  constexpr int HW  = H * W;
  __shared__ unsigned short A_sm[32 * AST];
  __shared__ float e_sm[32];

  const int tid  = threadIdx.x;
  const int lane = tid & 31;
  const int wave = tid >> 5;
  const int r0   = blockIdx.x * 32;
  const int b    = r0 / HW;
  const int rem  = r0 - b * HW;
  const int h    = rem / W;
  const int w0   = rem - h * W;          // 32 consecutive rows share (b,h)

  if (tid < 32) e_sm[tid] = 0.f;

  // Stage augmented A tile: [32 rows x KTOT] bf16 = [features | conv taps | 0pad]
  for (int idx = tid; idx < 32 * KTOT; idx += 256) {
    const int m = idx / KTOT;
    const int c = idx - m * KTOT;
    float v = 0.f;
    if (c < C) {
      v = inputs[(size_t)(r0 + m) * C + c];
    } else {
      const int t = c - C;
      if (t < T) {
        const int dy = t / KW - PAD;
        const int dx = (t % KW) - PAD;
        const int hh = h + dy;
        const int ww = w0 + m + dx;
        if (hh >= 0 && hh < H && ww >= 0 && ww < W)
          v = attn_sum[(size_t)b * HW + hh * W + ww];
      }
    }
    A_sm[m * AST + c] = f2bf(v);
  }
  __syncthreads();

  const int mtile = wave >> 2;           // row subtile 0..1
  const int ngrp  = wave & 3;            // column group 0..3
  const int ml    = lane & 15;           // A row within subtile == D column-in-tile
  const int half  = lane >> 4;
  const unsigned short* Arow = &A_sm[(mtile * 16 + ml) * AST];

  // Preload per-column-group scalars into registers (keeps the GEMM stream
  // free of interleaved b32 loads that would force loadcnt drains).
  float addv_a[8], w4_a[8];
#pragma unroll
  for (int i = 0; i < 8; ++i) {
    const int d = (ngrp + 4 * i) * 16 + ml;
    addv_a[i] = fb[d] + upred[(size_t)b * 512 + d];
    w4_a[i]   = w4[d];
  }

#pragma unroll
  for (int i = 0; i < 8; ++i) {
    const int ntile = ngrp + 4 * i;      // 0..31 (D = 512)
    const float addv = addv_a[i];
    const float w4v  = w4_a[i];
    v8f acc = {0.f, 0.f, 0.f, 0.f, 0.f, 0.f, 0.f, 0.f};
    const unsigned int* bbase = Bsw + ((size_t)ntile * NK * 32 + lane) * 8;

    for (int kt = 0; kt < NK; ++kt) {
      // A fragment (16x32 bf16): vgpr v holds k = half*16 + v*2, +1
      union { v16bf v; unsigned int u[8]; } Af;
#pragma unroll
      for (int vv = 0; vv < 8; ++vv) {
        const int kloc = (half << 4) + (vv << 1);
        Af.u[vv] = *reinterpret_cast<const unsigned int*>(Arow + kt * 32 + kloc);
      }
      // B fragment: pre-swizzled, 32 contiguous bytes per lane
      union { v16bf v; uint4 q[2]; } Bf;
      const unsigned int* bp = bbase + (size_t)kt * 256;
      Bf.q[0] = *reinterpret_cast<const uint4*>(bp);
      Bf.q[1] = *reinterpret_cast<const uint4*>(bp + 4);

      acc = __builtin_amdgcn_wmma_f32_16x16x32_bf16(
          false, Af.v, false, Bf.v, (short)0, acc, false, false);
    }

    // response = tanh(acc + upred + fb); e += response * w4  (reduce over N)
#pragma unroll
    for (int r = 0; r < 8; ++r) {
      float val = tanhf(acc[r] + addv) * w4v;
      val += __shfl_xor(val, 1);
      val += __shfl_xor(val, 2);
      val += __shfl_xor(val, 4);
      val += __shfl_xor(val, 8);
      if (ml == 0) atomicAdd(&e_sm[mtile * 16 + (half << 3) + r], val);
    }
  }
  __syncthreads();
  if (tid < 32) e_out[r0 + tid] = e_sm[tid];
}

// ---------------------------------------------------------------------------
// Softmax over H*W per batch (in-place e -> attn) + new_attn_sum to d_out
// ---------------------------------------------------------------------------
__global__ __launch_bounds__(256) void k_softmax_ns(
    float* __restrict__ e, const float* __restrict__ attn_sum,
    float* __restrict__ ns_out, int HW)
{
  __shared__ float red[256];
  const int b = blockIdx.x, tid = threadIdx.x;
  float* eb = e + (size_t)b * HW;
  const float* asb = attn_sum + (size_t)b * HW;
  float* nsb = ns_out + (size_t)b * HW;

  float mx = -3.402823e38f;
  for (int p = tid; p < HW; p += 256) mx = fmaxf(mx, eb[p]);
  red[tid] = mx; __syncthreads();
  for (int s = 128; s > 0; s >>= 1) { if (tid < s) red[tid] = fmaxf(red[tid], red[tid + s]); __syncthreads(); }
  mx = red[0]; __syncthreads();

  float sum = 0.f;
  for (int p = tid; p < HW; p += 256) sum += expf(eb[p] - mx);
  red[tid] = sum; __syncthreads();
  for (int s = 128; s > 0; s >>= 1) { if (tid < s) red[tid] += red[tid + s]; __syncthreads(); }
  const float inv = 1.f / red[0];

  for (int p = tid; p < HW; p += 256) {
    float a = expf(eb[p] - mx) * inv;
    eb[p] = a;
    nsb[p] = asb[p] + a;
  }
}

// ---------------------------------------------------------------------------
// context[b,c] = sum_p attn[b,p] * inputs[b,p,c]   (memory-bound reduction)
// ---------------------------------------------------------------------------
__global__ __launch_bounds__(256) void k_context(
    const float* __restrict__ attn, const float* __restrict__ inputs,
    float* __restrict__ ctx, int HW, int C, int cbase)
{
  __shared__ float red[256];
  const int gpb = C / 64;
  const int b = blockIdx.x / gpb;
  const int c0 = (blockIdx.x % gpb) * 64;
  const int tid = threadIdx.x;
  const int cl = tid & 63, pj = tid >> 6;
  float s = 0.f;
  for (int p = pj; p < HW; p += 4)
    s += attn[(size_t)b * HW + p] * inputs[((size_t)b * HW + p) * C + c0 + cl];
  red[tid] = s; __syncthreads();
  if (pj == 0)
    ctx[(size_t)b * 768 + cbase + c0 + cl] =
        red[cl] + red[64 + cl] + red[128 + cl] + red[192 + cl];
}

// ---------------------------------------------------------------------------
// Final: GRU2 -> new_hidden; out = emb + nh@d2 + ctx@d3; maxout; logits@d4
// ---------------------------------------------------------------------------
__global__ __launch_bounds__(256) void k_final(
    const float* __restrict__ emb, const float* __restrict__ pred,
    const float* __restrict__ ctx,
    const float* __restrict__ g2k, const float* __restrict__ g2r,
    const float* __restrict__ g2b,
    const float* __restrict__ d2w, const float* __restrict__ d3w,
    const float* __restrict__ d4w,
    float* __restrict__ logits_out, float* __restrict__ nh_out)
{
  __shared__ float nh_sm[16 * 256];
  __shared__ float out_sm[16 * 256];
  __shared__ float mo_sm[16 * 128];
  const int tid = threadIdx.x;
  for (int o = tid; o < 16 * 256; o += 256) {
    int i = o >> 8, j = o & 255;
    float xz = g2b[j], xr = g2b[256 + j], xh = g2b[512 + j];
    for (int c = 0; c < 768; ++c) {
      float x = ctx[i * 768 + c];
      xz += x * g2k[c * 768 + j];
      xr += x * g2k[c * 768 + 256 + j];
      xh += x * g2k[c * 768 + 512 + j];
    }
    float hz = g2b[768 + j], hr = g2b[768 + 256 + j], hh = g2b[768 + 512 + j];
    for (int c = 0; c < 256; ++c) {
      float hv = pred[i * 256 + c];
      hz += hv * g2r[c * 768 + j];
      hr += hv * g2r[c * 768 + 256 + j];
      hh += hv * g2r[c * 768 + 512 + j];
    }
    float z = sigmoidf(xz + hz), r = sigmoidf(xr + hr);
    float hc = tanhf(xh + r * hh);
    float nh = z * pred[o] + (1.f - z) * hc;
    nh_sm[o] = nh; nh_out[o] = nh;
  }
  __syncthreads();
  for (int o = tid; o < 16 * 256; o += 256) {
    int i = o >> 8, e = o & 255;
    float s = emb[o];
    for (int j = 0; j < 256; ++j) s += nh_sm[i * 256 + j] * d2w[j * 256 + e];
    for (int c = 0; c < 768; ++c) s += ctx[i * 768 + c] * d3w[c * 256 + e];
    out_sm[o] = s;
  }
  __syncthreads();
  for (int o = tid; o < 16 * 128; o += 256) {
    int i = o >> 7, m2 = o & 127;
    mo_sm[o] = fmaxf(out_sm[i * 256 + 2 * m2], out_sm[i * 256 + 2 * m2 + 1]);
  }
  __syncthreads();
  for (int o = tid; o < 16 * 128; o += 256) {
    int i = o >> 7, v = o & 127;
    float s = 0.f;
    for (int m = 0; m < 128; ++m) s += mo_sm[i * 128 + m] * d4w[m * 128 + v];
    logits_out[o] = s;
  }
}

// ---------------------------------------------------------------------------
extern "C" void kernel_launch(void* const* d_in, const int* in_sizes, int n_in,
                              void* d_out, int out_size, void* d_ws, size_t ws_size,
                              hipStream_t stream)
{
  (void)in_sizes; (void)n_in; (void)out_size; (void)ws_size;
  const int*   tokens   = (const int*)  d_in[0];
  const float* low_res  = (const float*)d_in[1];
  const float* high_res = (const float*)d_in[2];
  const float* h0       = (const float*)d_in[3];
  const float* as_low   = (const float*)d_in[4];
  const float* as_high  = (const float*)d_in[5];
  const float* emb_t    = (const float*)d_in[6];
  const float* g1k = (const float*)d_in[7];
  const float* g1r = (const float*)d_in[8];
  const float* g1b = (const float*)d_in[9];
  const float* g2k = (const float*)d_in[10];
  const float* g2r = (const float*)d_in[11];
  const float* g2b = (const float*)d_in[12];
  const float* d1w = (const float*)d_in[13];
  const float* d2w = (const float*)d_in[14];
  const float* d3w = (const float*)d_in[15];
  const float* d4w = (const float*)d_in[16];
  const float* cal_w1 = (const float*)d_in[17];
  const float* cal_b1 = (const float*)d_in[18];
  const float* cal_w2 = (const float*)d_in[19];
  const float* cal_w3 = (const float*)d_in[20];
  const float* cal_w4 = (const float*)d_in[21];
  const float* cah_w1 = (const float*)d_in[22];
  const float* cah_b1 = (const float*)d_in[23];
  const float* cah_w2 = (const float*)d_in[24];
  const float* cah_w3 = (const float*)d_in[25];
  const float* cah_w4 = (const float*)d_in[26];

  float* out     = (float*)d_out;
  float* logits  = out;            // 16*128
  float* nh_out  = out + 2048;     // 16*256
  float* ns_low  = out + 6144;     // 16*2048
  float* ns_high = out + 38912;    // 16*8192

  float* wsf     = (float*)d_ws;
  float* pred    = wsf;            // 4096
  float* emb     = wsf + 4096;     // 4096
  float* upred   = wsf + 8192;     // 8192
  float* fb_low  = wsf + 16384;    // 512
  float* fb_high = wsf + 16896;    // 512
  float* ctx     = wsf + 17408;    // 12288
  float* e_low   = wsf + 29696;    // 32768  (becomes attn_low in place)
  float* e_high  = wsf + 62464;    // 131072 (becomes attn_high in place)
  unsigned short* Bsw_low  = (unsigned short*)(wsf + 193536); // 640*512 bf16
  unsigned short* Bsw_high = Bsw_low + 640 * 512;             // 320*512 bf16

  k1_embed_gru1_upred<<<1, 256, 0, stream>>>(tokens, emb_t, h0, g1k, g1r, g1b,
                                             d1w, emb, pred, upred);
  k_prep_fb<<<2, 256, 0, stream>>>(cal_b1, cal_w2, fb_low);
  k_prep_fb<<<2, 256, 0, stream>>>(cah_b1, cah_w2, fb_high);
  k_prep_B<<<(640 * 512 + 255) / 256, 256, 0, stream>>>(cal_w1, cal_w2, cal_w3,
                                                        Bsw_low, 512, 121, 640);
  k_prep_B<<<(320 * 512 + 255) / 256, 256, 0, stream>>>(cah_w1, cah_w2, cah_w3,
                                                        Bsw_high, 256, 49, 320);

  k_attn_scores<512, 121, 11, 5, 32, 64, 640>
      <<<(16 * 32 * 64) / 32, 256, 0, stream>>>(
          low_res, as_low, upred, fb_low, cal_w4,
          (const unsigned int*)Bsw_low, e_low);
  k_attn_scores<256, 49, 7, 3, 64, 128, 320>
      <<<(16 * 64 * 128) / 32, 256, 0, stream>>>(
          high_res, as_high, upred, fb_high, cah_w4,
          (const unsigned int*)Bsw_high, e_high);

  k_softmax_ns<<<16, 256, 0, stream>>>(e_low, as_low, ns_low, 2048);
  k_softmax_ns<<<16, 256, 0, stream>>>(e_high, as_high, ns_high, 8192);

  k_context<<<16 * 8, 256, 0, stream>>>(e_low, low_res, ctx, 2048, 512, 0);
  k_context<<<16 * 4, 256, 0, stream>>>(e_high, high_res, ctx, 8192, 256, 512);

  k_final<<<1, 256, 0, stream>>>(emb, pred, ctx, g2k, g2r, g2b,
                                 d2w, d3w, d4w, logits, nh_out);
}
